// TokenGNNEncoderLayer_68650757259662
// MI455X (gfx1250) — matrix operand order repaired and checked by reading, not compiled, not run on hardware
//
#include <hip/hip_runtime.h>
#include <hip/hip_bf16.h>
#include <math.h>

// ---------------- problem constants (match reference) ----------------
#define NND  100000      // nodes
#define E0C  160000      // edges per mask segment
#define EC   800000      // total edges
#define DC   128         // embed dim
#define HC   8           // heads
#define DKC  16          // head dim
#define NGC  50          // gaussians
#define SCALEC 0.25f     // 1/sqrt(16)
#define EPSC 1e-5f

typedef _Float16 v16h __attribute__((ext_vector_type(16)));
typedef _Float16 v8h  __attribute__((ext_vector_type(8)));
typedef _Float16 v4h  __attribute__((ext_vector_type(4)));
typedef float    v8f  __attribute__((ext_vector_type(8)));

// ---------------- helpers ----------------
__device__ __forceinline__ float waveRedSum(float v) {
#pragma unroll
  for (int m = 16; m >= 1; m >>= 1) v += __shfl_xor(v, m, 32);
  return v;
}

// native global_atomic_add_f32 (no-return) — coarse-grained HBM is safe here
__device__ __forceinline__ void atomicAddF(float* p, float v) {
  unsafeAtomicAdd(p, v);
}

// float atomic-max via monotonic int/uint trick (requires init to -inf)
__device__ __forceinline__ void atomicMaxF(float* p, float v) {
  if (v >= 0.0f) atomicMax((int*)p, __float_as_int(v));
  else           atomicMin((unsigned int*)p, __float_as_uint(v));
}

__device__ __forceinline__ float gelu_tanh(float x) {
  return 0.5f * x * (1.0f + tanhf(0.7978845608028654f * (x + 0.044715f * x * x * x)));
}

// ---------------- 0) per-call state init ----------------
__global__ __launch_bounds__(256)
void init_state(float* __restrict__ agg, float* __restrict__ denom,
                float* __restrict__ segmax) {
  const int NAGG = NND * DC, NSEG = NND * HC;
  int t = blockIdx.x * 256 + threadIdx.x;
  if (t < NAGG) agg[t] = 0.0f;
  else if (t < NAGG + NSEG) denom[t - NAGG] = 0.0f;
  else if (t < NAGG + 2 * NSEG) segmax[t - NAGG - NSEG] = -INFINITY;
}

// ---------------- 1) f32 weight [in,out] -> f16 transposed [out][in] ----------------
__global__ __launch_bounds__(256)
void transpose_to_f16(const float* __restrict__ src, _Float16* __restrict__ dst,
                      int R, int C) {  // src is R x C (row-major), dst is C x R
  int t = blockIdx.x * 256 + threadIdx.x;
  if (t >= R * C) return;
  int i = t / C, o = t - i * C;
  dst[(size_t)o * R + i] = (_Float16)src[t];
}

// ---------------- 2) LayerNorm (f32 in) -> f16 out, one wave per row ----------------
__global__ __launch_bounds__(256)
void ln_f16(const float* __restrict__ x, const float* __restrict__ gam,
            const float* __restrict__ bet, _Float16* __restrict__ out) {
  int wave = threadIdx.x >> 5, lane = threadIdx.x & 31;
  int row = blockIdx.x * 8 + wave;
  if (row >= NND) return;
  const float4 xv = *(const float4*)(x + (size_t)row * DC + lane * 4);
  float mu = waveRedSum(xv.x + xv.y + xv.z + xv.w) * (1.0f / DC);
  float d0 = xv.x - mu, d1 = xv.y - mu, d2 = xv.z - mu, d3 = xv.w - mu;
  float var = waveRedSum(d0 * d0 + d1 * d1 + d2 * d2 + d3 * d3) * (1.0f / DC);
  float inv = rsqrtf(var + EPSC);
  const float4 gv = *(const float4*)(gam + lane * 4);
  const float4 bv = *(const float4*)(bet + lane * 4);
  v4h o;
  o[0] = (_Float16)(d0 * inv * gv.x + bv.x);
  o[1] = (_Float16)(d1 * inv * gv.y + bv.y);
  o[2] = (_Float16)(d2 * inv * gv.z + bv.z);
  o[3] = (_Float16)(d3 * inv * gv.w + bv.w);
  *(v4h*)(out + (size_t)row * DC + lane * 4) = o;
}

// ---------------- 3) WMMA GEMM ----------------
// Each wave computes TWO adjacent 16x16 output tiles sharing one A fragment:
// halves activation traffic and gives 2 v_wmma per A-load pair.
// A: [N x K] f16 row-major.  Bt: [NCOL x K] f16 (pre-transposed weights).
// mode 0: outF = acc + bias
// mode 1: outF = acc + bias + resid   (residual add)
// mode 2: outH = gelu(acc + bias)     (f16 output for next GEMM)
__global__ __launch_bounds__(128)
void wmma_gemm(const _Float16* __restrict__ A, const _Float16* __restrict__ Bt,
               const float* __restrict__ bias, const float* __restrict__ resid,
               float* __restrict__ outF, _Float16* __restrict__ outH,
               int K, int NCOL, int mode) {
  const int wave = threadIdx.x >> 5;
  const int lane = threadIdx.x & 31;
  const int g = lane >> 4;       // lane group: selects K half per ISA layout
  const int n = lane & 15;       // row (A) / col (B,D) within tile
  const int r0 = blockIdx.x * 16;
  const int c0 = (blockIdx.y * 8 + wave * 2) * 16;   // first of two col tiles

  const _Float16* arow  = A  + (size_t)(r0 + n) * K;
  const _Float16* brow0 = Bt + (size_t)(c0 + n) * K;
  const _Float16* brow1 = Bt + (size_t)(c0 + 16 + n) * K;

  v8f acc0 = {}, acc1 = {};
  const int kblocks = K >> 5;
#pragma unroll 4
  for (int kb = 0; kb < kblocks; ++kb) {
    // A 16x32 f16 fragment: lane holds K = [8g,8g+8) and [16+8g,16+8g+8)
    const _Float16* ab = arow + kb * 32;
    v8h alo = *(const v8h*)(ab + 8 * g);
    v8h ahi = *(const v8h*)(ab + 16 + 8 * g);
    // B 32x16 f16 fragments: lane holds K = [8g,8g+8) and [16+8g,16+8g+8)
    const _Float16* bb0 = brow0 + kb * 32 + 8 * g;
    const _Float16* bb1 = brow1 + kb * 32 + 8 * g;
    v8h b0lo = *(const v8h*)(bb0);
    v8h b0hi = *(const v8h*)(bb0 + 16);
    v8h b1lo = *(const v8h*)(bb1);
    v8h b1hi = *(const v8h*)(bb1 + 16);
    v16h af, bf0, bf1;
#pragma unroll
    for (int i = 0; i < 8; ++i) {
      af[i]  = alo[i];  af[i + 8]  = ahi[i];
      bf0[i] = b0lo[i]; bf0[i + 8] = b0hi[i];
      bf1[i] = b1lo[i]; bf1[i + 8] = b1hi[i];
    }
    acc0 = __builtin_amdgcn_wmma_f32_16x16x32_f16(
        false, af, false, bf0, (short)0, acc0, false, false);
    acc1 = __builtin_amdgcn_wmma_f32_16x16x32_f16(
        false, af, false, bf1, (short)0, acc1, false, false);
  }

  // D layout: element r -> row r0 + r + 8g, col = tile_base + n
  const int col0 = c0 + n, col1 = c0 + 16 + n;
  const float bc0 = bias[col0], bc1 = bias[col1];
#pragma unroll
  for (int r = 0; r < 8; ++r) {
    const int row = r0 + r + 8 * g;
    float v0 = acc0[r] + bc0;
    float v1 = acc1[r] + bc1;
    if (mode == 2) {
      outH[(size_t)row * NCOL + col0] = (_Float16)gelu_tanh(v0);
      outH[(size_t)row * NCOL + col1] = (_Float16)gelu_tanh(v1);
    } else {
      if (mode == 1) {
        v0 += resid[(size_t)row * NCOL + col0];
        v1 += resid[(size_t)row * NCOL + col1];
      }
      outF[(size_t)row * NCOL + col0] = v0;
      outF[(size_t)row * NCOL + col1] = v1;
    }
  }
}

// ---------------- 4) gaussian smear + per-segment linear -> attn_mask [E,H] --------
__global__ __launch_bounds__(256)
void gauss_mask(const float* __restrict__ ad, const float* __restrict__ dd,
                const float* __restrict__ cdd, const float* __restrict__ css,
                const float* Wnn, const float* bnn, const float* Wne, const float* bne,
                const float* Wen, const float* ben, const float* Wdd, const float* bdd,
                const float* Wss, const float* bss, float* __restrict__ mask) {
  int e = blockIdx.x * 256 + threadIdx.x;
  if (e >= EC) return;
  int s = e / E0C, r = e - s * E0C;
  const float *W, *b;
  float dv, start, stop;
  switch (s) {
    case 0:  W = Wnn; b = bnn; dv = ad[r];  start = 0.f;  stop = 12.f; break;
    case 1:  W = Wne; b = bne; dv = dd[r];  start = 0.f;  stop = 12.f; break;
    case 2:  W = Wen; b = ben; dv = dd[r];  start = 0.f;  stop = 12.f; break;
    case 3:  W = Wdd; b = bdd; dv = cdd[r]; start = -1.f; stop = 1.f;  break;
    default: W = Wss; b = bss; dv = css[r]; start = -1.f; stop = 1.f;  break;
  }
  const float step  = (stop - start) * (1.0f / (NGC - 1));
  const float coeff = -0.5f / (step * step);
  float acc[8];
#pragma unroll
  for (int h = 0; h < 8; ++h) acc[h] = b[h];
  for (int gi = 0; gi < NGC; ++gi) {
    float d = dv - (start + gi * step);
    float w = __expf(coeff * d * d);
    const float4 w0 = *(const float4*)(W + gi * 8);
    const float4 w1 = *(const float4*)(W + gi * 8 + 4);
    acc[0] += w * w0.x; acc[1] += w * w0.y; acc[2] += w * w0.z; acc[3] += w * w0.w;
    acc[4] += w * w1.x; acc[5] += w * w1.y; acc[6] += w * w1.z; acc[7] += w * w1.w;
  }
  float4 o0 = {acc[0], acc[1], acc[2], acc[3]};
  float4 o1 = {acc[4], acc[5], acc[6], acc[7]};
  *(float4*)(mask + (size_t)e * 8) = o0;
  *(float4*)(mask + (size_t)e * 8 + 4) = o1;
}

// ---------------- 5) pass A: logits = scale*q.kv + mask; segment max ----------------
__global__ __launch_bounds__(256)
void edge_logits(const float* __restrict__ zk, const int* __restrict__ edges,
                 float* __restrict__ logits, float* __restrict__ segmax) {
  int t = blockIdx.x * 256 + threadIdx.x;
  if (t >= EC * HC) return;
  int e = t >> 3, h = t & 7;
  int src = edges[e], dst = edges[EC + e];
  const float4* q = (const float4*)(zk + (size_t)src * DC + h * DKC);
  const float4* k = (const float4*)(zk + (size_t)dst * DC + h * DKC);
  float dot = 0.0f;
#pragma unroll
  for (int i = 0; i < 4; ++i) {
    float4 a = q[i], b = k[i];
    dot += a.x * b.x + a.y * b.y + a.z * b.z + a.w * b.w;
  }
  float l = SCALEC * dot + logits[t];   // logits buffer pre-holds the gaussian mask
  logits[t] = l;
  atomicMaxF(&segmax[(size_t)src * HC + h], l);
}

// ---------------- 6) pass B: exp, denom sum, unnormalized scatter-aggregate --------
__global__ __launch_bounds__(256)
void edge_scatter(const float* __restrict__ zk, const int* __restrict__ edges,
                  const float* __restrict__ logits, const float* __restrict__ segmax,
                  float* __restrict__ denom, float* __restrict__ agg) {
  int t = blockIdx.x * 256 + threadIdx.x;
  if (t >= EC * HC) return;
  int e = t >> 3, h = t & 7;
  int src = edges[e], dst = edges[EC + e];
  float m = segmax[(size_t)src * HC + h];
  float s = __expf(logits[t] - m);
  atomicAddF(&denom[(size_t)src * HC + h], s);
  const float4* k = (const float4*)(zk + (size_t)dst * DC + h * DKC);
  float* a = agg + (size_t)src * DC + h * DKC;
#pragma unroll
  for (int i = 0; i < 4; ++i) {
    float4 kv = k[i];
    atomicAddF(a + 4 * i + 0, s * kv.x);
    atomicAddF(a + 4 * i + 1, s * kv.y);
    atomicAddF(a + 4 * i + 2, s * kv.z);
    atomicAddF(a + 4 * i + 3, s * kv.w);
  }
}

// ---------------- 7) normalize aggregate -> f16 for the Wo GEMM ----------------
__global__ __launch_bounds__(256)
void finalize_agg(const float* __restrict__ agg, const float* __restrict__ denom,
                  _Float16* __restrict__ aggh) {
  int t = blockIdx.x * 256 + threadIdx.x;
  if (t >= NND * DC) return;
  int n = t >> 7, c = t & 127, h = c >> 4;
  float den = denom[(size_t)n * HC + h];
  float v = (den > 0.0f) ? agg[t] / den : 0.0f;
  aggh[t] = (_Float16)v;
}

// ---------------- host-side launch ----------------
extern "C" void kernel_launch(void* const* d_in, const int* in_sizes, int n_in,
                              void* d_out, int out_size, void* d_ws, size_t ws_size,
                              hipStream_t stream) {
  (void)in_sizes; (void)n_in; (void)out_size; (void)ws_size;
  const float* x        = (const float*)d_in[0];
  const int*   edges    = (const int*)  d_in[1];
  const float* all_dist = (const float*)d_in[2];
  const float* dist     = (const float*)d_in[3];
  const float* cdd      = (const float*)d_in[4];
  const float* css      = (const float*)d_in[5];
  const float* Wk  = (const float*)d_in[6];   const float* bk  = (const float*)d_in[7];
  const float* Wnn = (const float*)d_in[8];   const float* bnn = (const float*)d_in[9];
  const float* Wne = (const float*)d_in[10];  const float* bne = (const float*)d_in[11];
  const float* Wen = (const float*)d_in[12];  const float* ben = (const float*)d_in[13];
  const float* Wdd = (const float*)d_in[14];  const float* bdd = (const float*)d_in[15];
  const float* Wss = (const float*)d_in[16];  const float* bss = (const float*)d_in[17];
  const float* Wo  = (const float*)d_in[18];  const float* bo  = (const float*)d_in[19];
  const float* ln1g = (const float*)d_in[20]; const float* ln1b = (const float*)d_in[21];
  const float* ln2g = (const float*)d_in[22]; const float* ln2b = (const float*)d_in[23];
  const float* W1  = (const float*)d_in[24];  const float* bf1 = (const float*)d_in[25];
  const float* W2  = (const float*)d_in[26];  const float* bf2 = (const float*)d_in[27];
  float* out = (float*)d_out;

  // workspace layout (bytes, all 256B aligned); buffers aliased across phases
  char* ws = (char*)d_ws;
  _Float16* zh     = (_Float16*)(ws + 0);           // 25.6 MB  (later: aggh)
  float*    zk     = (float*)   (ws + 25600000);    // 51.2 MB  (later: ffh f16 N*256)
  float*    logits = (float*)   (ws + 76800000);    // 25.6 MB  (later: z2h f16)
  float*    segmax = (float*)   (ws + 102400000);   //  3.2 MB
  float*    denom  = (float*)   (ws + 105600000);   //  3.2 MB
  float*    agg    = (float*)   (ws + 108800000);   // 51.2 MB
  _Float16* wkT    = (_Float16*)(ws + 160000000);
  _Float16* woT    = wkT + 128 * 128;
  _Float16* w1T    = woT + 128 * 128;
  _Float16* w2T    = w1T + 128 * 256;
  _Float16* aggh   = zh;
  _Float16* z2h    = (_Float16*)logits;
  _Float16* ffh    = (_Float16*)zk;

  // 0) re-init atomic state every call (graph-replay safe)
  {
    int tot = NND * DC + 2 * NND * HC;
    init_state<<<(tot + 255) / 256, 256, 0, stream>>>(agg, denom, segmax);
  }
  // 1) weights -> transposed f16 panels
  transpose_to_f16<<<(128 * 128) / 256, 256, 0, stream>>>(Wk, wkT, 128, 128);
  transpose_to_f16<<<(128 * 128) / 256, 256, 0, stream>>>(Wo, woT, 128, 128);
  transpose_to_f16<<<(128 * 256) / 256, 256, 0, stream>>>(W1, w1T, 128, 256);
  transpose_to_f16<<<(256 * 128) / 256, 256, 0, stream>>>(W2, w2T, 256, 128);
  // 2) z = LN1(x) -> f16
  ln_f16<<<NND / 8, 256, 0, stream>>>(x, ln1g, ln1b, zh);
  // 3) zk = z @ Wk + bk   (WMMA; one block covers all 128 cols)
  wmma_gemm<<<dim3(NND / 16, 1), 128, 0, stream>>>(zh, wkT, bk, nullptr, zk, nullptr,
                                                   128, 128, 0);
  // 4) gaussian edge mask -> logits buffer
  gauss_mask<<<EC / 256, 256, 0, stream>>>(all_dist, dist, cdd, css,
                                           Wnn, bnn, Wne, bne, Wen, ben,
                                           Wdd, bdd, Wss, bss, logits);
  // 5) logits += scale*q.kv ; per-segment max
  edge_logits<<<(EC * HC) / 256, 256, 0, stream>>>(zk, edges, logits, segmax);
  // 6) exp / denom / unnormalized aggregate
  edge_scatter<<<(EC * HC) / 256, 256, 0, stream>>>(zk, edges, logits, segmax,
                                                    denom, agg);
  // 7) agg / denom -> f16
  finalize_agg<<<(NND * DC) / 256, 256, 0, stream>>>(agg, denom, aggh);
  // 8) x2 = x + agg @ Wo + bo   -> d_out
  wmma_gemm<<<dim3(NND / 16, 1), 128, 0, stream>>>(aggh, woT, bo, x, out, nullptr,
                                                   128, 128, 1);
  // 9) z2 = LN2(x2) -> f16
  ln_f16<<<NND / 8, 256, 0, stream>>>(out, ln2g, ln2b, z2h);
  // 10) ff1 = gelu(z2 @ W1 + bf1) -> f16 [N,256]
  wmma_gemm<<<dim3(NND / 16, 2), 128, 0, stream>>>(z2h, w1T, bf1, nullptr, nullptr, ffh,
                                                   128, 256, 2);
  // 11) out = x2 + ff1 @ W2 + bf2   (in-place residual on d_out)
  wmma_gemm<<<dim3(NND / 16, 1), 128, 0, stream>>>(ffh, w2T, bf2, out, out, nullptr,
                                                   256, 128, 1);
}
